// GrapherSetEdges_36910948941896
// MI455X (gfx1250) — matrix operand order repaired and checked by reading, not compiled, not run on hardware
//
#include <hip/hip_runtime.h>

// ---------------- problem constants ----------------
#define N_NODES 2048
#define C_CH    16
#define PX      64        // 8*8 pixels
#define F_DIM   1024      // C*HW*HW
#define HEADS   4
#define HF      4096      // HEADS*F
#define E_EDGES 16384
#define EN      18432     // E + N (self loops)
#define EPS     1e-5f
#define SLOPE   0.2f

typedef __bf16 bhalf;
typedef __attribute__((ext_vector_type(16))) __bf16 v16bf;
typedef __attribute__((ext_vector_type(8)))  __bf16 v8bf;
typedef __attribute__((ext_vector_type(8)))  float  v8f;
typedef __attribute__((ext_vector_type(4)))  unsigned int u32x4;
typedef __attribute__((ext_vector_type(8)))  int i32x8;
typedef __attribute__((ext_vector_type(4)))  int i32x4;

#if defined(__HIP_DEVICE_COMPILE__) && __has_builtin(__builtin_amdgcn_tensor_load_to_lds) && __has_builtin(__builtin_amdgcn_s_wait_tensorcnt)
#define USE_TDM 1
#else
#define USE_TDM 0
#endif

static __device__ __forceinline__ unsigned short f2bf(float f) {
  unsigned int u = __float_as_uint(f);
  unsigned int r = u + 0x7FFFu + ((u >> 16) & 1u);   // round-to-nearest-even
  return (unsigned short)(r >> 16);
}

#if USE_TDM
// Build a 2D TDM descriptor (D#) per CDNA5 ISA 8.3-8.6 and issue TENSOR_LOAD_TO_LDS.
// data_size = 2 bytes (bf16). Dims/strides in elements. Groups 2/3 (and the trailing
// group of the 6-arg builtin) zero: 2D tile, tile_dim2 = 0 marks higher dims unused.
static __device__ __forceinline__ void tdm_load_2d(unsigned lds_off, const void* gptr,
                                                   unsigned tensor_d0, unsigned tensor_d1,
                                                   unsigned tile_d0, unsigned tile_d1,
                                                   unsigned stride0) {
  unsigned long long ga = (unsigned long long)(size_t)gptr;
  u32x4 g0;
  g0[0] = 1u;                                           // count=1, user mode
  g0[1] = lds_off;                                      // lds_addr (bytes)
  g0[2] = (unsigned)(ga & 0xFFFFFFFFu);                 // global_addr[31:0]
  g0[3] = (unsigned)((ga >> 32) & 0x1FFFFFFu) | (2u << 30);  // global_addr[56:32] | type=2
  i32x8 g1;
  g1[0] = (int)(1u << 16);                              // data_size=1 -> 2 bytes
  g1[1] = (int)((tensor_d0 & 0xFFFFu) << 16);           // tensor_dim0[15:0] at bits[63:48]
  g1[2] = (int)((tensor_d0 >> 16) | ((tensor_d1 & 0xFFFFu) << 16));
  g1[3] = (int)((tensor_d1 >> 16) | (tile_d0 << 16));   // tensor_dim1[31:16] | tile_dim0
  g1[4] = (int)tile_d1;                                 // tile_dim1 | tile_dim2(=0)<<16
  g1[5] = (int)stride0;                                 // tensor_dim0_stride[31:0]
  g1[6] = 0;                                            // stride0[47:32] | stride1[15:0]
  g1[7] = 0;                                            // stride1[47:16]
  i32x4 z4 = {};
  i32x8 z8 = {};
  __builtin_amdgcn_tensor_load_to_lds(g0, g1, z4, z4, z8, 0);   // 6-arg toolchain form
}
#endif

// ---------------- fp32 -> bf16 weight conversion ----------------
__global__ void cvt_bf16_kernel(const float* __restrict__ src,
                                unsigned short* __restrict__ dst, int n) {
  int i = blockIdx.x * 256 + threadIdx.x;
  if (i < n) dst[i] = f2bf(src[i]);
}

// ---------------- fc1: 1x1 conv 16->16, block = one image, thread = one pixel ----------------
__global__ void fc1_kernel(const float* __restrict__ x, const float* __restrict__ w1,
                           const float* __restrict__ b1, float* __restrict__ y1) {
  __shared__ float ws[C_CH * C_CH];
  __shared__ float bs[C_CH];
  int n = blockIdx.x, t = threadIdx.x;
  for (int i = t; i < C_CH * C_CH; i += 64) ws[i] = w1[i];
  if (t < C_CH) bs[t] = b1[t];
  __syncthreads();
  const float* xp = x + (size_t)n * F_DIM + t;
  float xv[C_CH];
#pragma unroll
  for (int c = 0; c < C_CH; ++c) xv[c] = xp[c * PX];
  float* yp = y1 + (size_t)n * F_DIM + t;
#pragma unroll
  for (int o = 0; o < C_CH; ++o) {
    float a = bs[o];
#pragma unroll
    for (int c = 0; c < C_CH; ++c) a = fmaf(ws[o * C_CH + c], xv[c], a);
    yp[o * PX] = a;
  }
}

// ---------------- BN stats+finalize: one block per channel (deterministic, no atomics) ----------------
__global__ void bn_stats_kernel(const float* __restrict__ y, const float* __restrict__ gamma,
                                const float* __restrict__ beta, float* __restrict__ scale,
                                float* __restrict__ shift) {
  int c = blockIdx.x, t = threadIdx.x;
  __shared__ float s0[256], s1[256];
  float a = 0.f, b = 0.f;
  for (int i = t; i < N_NODES * PX; i += 256) {
    int n = i >> 6, p = i & 63;
    float v = y[(size_t)n * F_DIM + c * PX + p];
    a += v; b += v * v;
  }
  s0[t] = a; s1[t] = b; __syncthreads();
  for (int o = 128; o > 0; o >>= 1) {
    if (t < o) { s0[t] += s0[t + o]; s1[t] += s1[t + o]; }
    __syncthreads();
  }
  if (t == 0) {
    float inv = 1.f / (float)(N_NODES * PX);
    float m = s0[0] * inv, v = s1[0] * inv - m * m;
    float sc = gamma[c] * rsqrtf(v + EPS);
    scale[c] = sc; shift[c] = beta[c] - m * sc;
  }
}

// ---------------- apply BN1 and convert node features to bf16 ----------------
__global__ void bn_apply_cvt_kernel(const float* __restrict__ y, const float* __restrict__ scale,
                                    const float* __restrict__ shift,
                                    unsigned short* __restrict__ o) {
  int i = blockIdx.x * 256 + threadIdx.x;     // exactly N*F threads
  int c = (i >> 6) & 15;
  o[i] = f2bf(fmaf(y[i], scale[c], shift[c]));
}

// ---------------- bf16 WMMA GEMM: [2048,1024] x [1024,4096] + bias -> fp32 ----------------
// block = 128x128 tile, 8 waves; wave = 32(M) x 64(N) = 2x4 wmma tiles; K-step 32.
// TDM path: wave 0 DMAs each 128x32 / 32x128 tile into double-buffered LDS
// (TENSOR_LOAD_TO_LDS, TENSORcnt) so the copy of tile k+1 overlaps WMMA of tile k.
__global__ __launch_bounds__(256)
void gemm_bf16_wmma_kernel(const unsigned short* __restrict__ A,   // [2048,1024] bf16
                           const unsigned short* __restrict__ B,   // [1024,4096] bf16
                           const float* __restrict__ bias,         // [4096]
                           float* __restrict__ Cout) {              // [2048,4096] f32
#if USE_TDM
  __shared__ __align__(32) unsigned short At[2][128 * 32];
  __shared__ __align__(32) unsigned short Bt[2][32 * 128];
#else
  __shared__ __align__(32) unsigned short At[1][128 * 32];
  __shared__ __align__(32) unsigned short Bt[1][32 * 128];
#endif
  const int t = threadIdx.x;
  const int lane = t & 31, wid = t >> 5;
  const int wm = wid & 3, wn = wid >> 2;      // wave tile: rows wm*32.., cols wn*64..
  const int mb = blockIdx.y * 128, nb = blockIdx.x * 128;

  v8f acc[2][4] = {};

#if USE_TDM
  if (wid == 0) {
    tdm_load_2d((unsigned)(size_t)(void*)At[0], A + (size_t)mb * F_DIM,
                F_DIM, N_NODES, 32, 128, F_DIM);
    tdm_load_2d((unsigned)(size_t)(void*)Bt[0], B + nb,
                HF, F_DIM, 128, 32, HF);
  }
#endif

  for (int it = 0; it < F_DIM / 32; ++it) {
    const int kk = it * 32;
#if USE_TDM
    const int cur = it & 1;
    if (wid == 0) {
      if (it + 1 < F_DIM / 32) {
        const int nk = kk + 32, nxt = (it + 1) & 1;
        tdm_load_2d((unsigned)(size_t)(void*)At[nxt], A + (size_t)mb * F_DIM + nk,
                    F_DIM, N_NODES, 32, 128, F_DIM);
        tdm_load_2d((unsigned)(size_t)(void*)Bt[nxt], B + (size_t)nk * HF + nb,
                    HF, F_DIM, 128, 32, HF);
        __builtin_amdgcn_s_wait_tensorcnt(2);   // tile `it` landed; 2 newest still in flight
      } else {
        __builtin_amdgcn_s_wait_tensorcnt(0);
      }
    }
    __syncthreads();
#else
    const int cur = 0;
#pragma unroll
    for (int j = 0; j < 2; ++j) {             // 2 x 16B per thread per tile
      int q = t + 256 * j;
      int ar = q >> 2, ac = (q & 3) * 8;      // A tile 128x32
      *(uint4*)(At[0] + ar * 32 + ac) =
          *(const uint4*)(A + (size_t)(mb + ar) * F_DIM + kk + ac);
      int br = q >> 4, bc = (q & 15) * 8;     // B tile 32x128
      *(uint4*)(Bt[0] + br * 128 + bc) =
          *(const uint4*)(B + (size_t)(kk + br) * HF + nb + bc);
    }
    if (kk + 32 < F_DIM) {                    // gfx1250 global_prefetch_b8 for next K tile
      __builtin_prefetch(A + (size_t)(mb + (t >> 1)) * F_DIM + kk + 32, 0, 1);
      __builtin_prefetch(B + (size_t)(kk + 32 + (t >> 3)) * HF + nb, 0, 1);
    }
    __syncthreads();
#endif

    // A fragments: lane l -> row m = l&15, K-half (l>>4)*8; elem i -> k=(i<8?i:i+8)
    union AF { struct { v8bf lo, hi; } p; v16bf v; } af[2];
    const bhalf* Abase = (const bhalf*)At[cur];
    const bhalf* Bbase = (const bhalf*)Bt[cur];
#pragma unroll
    for (int mt = 0; mt < 2; ++mt) {
      const bhalf* ap = Abase + (wm * 32 + mt * 16 + (lane & 15)) * 32 + (lane >> 4) * 8;
      af[mt].p.lo = *(const v8bf*)ap;
      af[mt].p.hi = *(const v8bf*)(ap + 16);
    }
    // B fragments: lane l -> col n = l&15, K-half (l>>4)*8; elem i -> k=(i<8?i:i+8)
#pragma unroll
    for (int nt = 0; nt < 4; ++nt) {
      v16bf bfrag;
#pragma unroll
      for (int s = 0; s < 16; ++s)
        bfrag[s] = Bbase[((lane >> 4) * 8 + (s < 8 ? s : s + 8)) * 128 + wn * 64 + nt * 16 + (lane & 15)];
#pragma unroll
      for (int mt = 0; mt < 2; ++mt)
        acc[mt][nt] = __builtin_amdgcn_wmma_f32_16x16x32_bf16(
            false, af[mt].v, false, bfrag, (short)0, acc[mt][nt], false, false);
    }
    __syncthreads();
  }

  // C layout: lane l -> col l&15, VGPR r -> row r + 8*(l>>4)
  const int mrow = mb + wm * 32, ncol = nb + wn * 64;
  const int cn = lane & 15, moff = (lane >> 4) * 8;
#pragma unroll
  for (int nt = 0; nt < 4; ++nt) {
    float bv = bias[ncol + nt * 16 + cn];
#pragma unroll
    for (int mt = 0; mt < 2; ++mt) {
#pragma unroll
      for (int r = 0; r < 8; ++r)
        Cout[(size_t)(mrow + mt * 16 + moff + r) * HF + ncol + nt * 16 + cn] =
            acc[mt][nt][r] + bv;
    }
  }
}

// ---------------- GAT init: out = bias_gat broadcast; mx = -inf; denom = 0 ----------------
__global__ void gat_init_kernel(float* __restrict__ gout, const float* __restrict__ bias) {
  int i = blockIdx.x * 256 + threadIdx.x;     // exactly N*HF threads
  gout[i] = bias[i & (HF - 1)];
}
__global__ void softmax_init_kernel(float* __restrict__ mx, float* __restrict__ denom) {
  int i = blockIdx.x * 256 + threadIdx.x;
  if (i < N_NODES * HEADS) { mx[i] = -__builtin_inff(); denom[i] = 0.f; }
}

// ---------------- edge logits: wave per edge; lane covers 128 contiguous feats (head = lane>>3) ----------------
__global__ __launch_bounds__(256)
void edge_logits_kernel(const float* __restrict__ xl, const float* __restrict__ xr,
                        const float* __restrict__ att, const int* __restrict__ ei,
                        float* __restrict__ logits, float* __restrict__ mx) {
  int e = blockIdx.x * 8 + (threadIdx.x >> 5);
  if (e >= EN) return;
  int lane = threadIdx.x & 31;
  int src, dst; bool valid;
  if (e < E_EDGES) { src = ei[e]; dst = ei[E_EDGES + e]; valid = (src != dst); }
  else             { src = dst = e - E_EDGES; valid = true; }
  if (!valid) {
    if (lane < HEADS) logits[e * HEADS + lane] = -__builtin_inff();
    return;
  }
  const float4* a = (const float4*)(xl + (size_t)src * HF) + lane * 32;
  const float4* b = (const float4*)(xr + (size_t)dst * HF) + lane * 32;
  const float4* w = (const float4*)att + lane * 32;   // att flat [h*1024+f] aligns with feature idx
  float acc = 0.f;
#pragma unroll 8
  for (int i = 0; i < 32; ++i) {
    float4 av = a[i], bv = b[i], wv = w[i]; float s;
    s = av.x + bv.x; acc = fmaf(s > 0.f ? s : SLOPE * s, wv.x, acc);
    s = av.y + bv.y; acc = fmaf(s > 0.f ? s : SLOPE * s, wv.y, acc);
    s = av.z + bv.z; acc = fmaf(s > 0.f ? s : SLOPE * s, wv.z, acc);
    s = av.w + bv.w; acc = fmaf(s > 0.f ? s : SLOPE * s, wv.w, acc);
  }
  acc += __shfl_xor(acc, 1);                  // reduce the 8 lanes of each head
  acc += __shfl_xor(acc, 2);
  acc += __shfl_xor(acc, 4);
  if ((lane & 7) == 0) {
    int h = lane >> 3;
    logits[e * HEADS + h] = acc;
    int* p = (int*)(mx + dst * HEADS + h);    // bit-trick float atomic max
    if (acc >= 0.f) atomicMax(p, __float_as_int(acc));
    else            atomicMin((unsigned int*)p, __float_as_uint(acc));
  }
}

// ---------------- exp + denom accumulation ----------------
__global__ void edge_expsum_kernel(const float* __restrict__ logits, const int* __restrict__ ei,
                                   const float* __restrict__ mx, float* __restrict__ denom,
                                   float* __restrict__ expv) {
  int i = blockIdx.x * 256 + threadIdx.x;
  if (i >= EN * HEADS) return;
  int e = i >> 2, h = i & 3;
  float lg = logits[i];
  if (lg == -__builtin_inff()) { expv[i] = 0.f; return; }
  int dst = (e < E_EDGES) ? ei[E_EDGES + e] : (e - E_EDGES);
  float w = __expf(lg - mx[dst * HEADS + h]);
  expv[i] = w;
  atomicAdd(denom + dst * HEADS + h, w);
}

// ---------------- weighted scatter: out[dst] += alpha * xl[src] ----------------
__global__ __launch_bounds__(256)
void edge_scatter_kernel(const float* __restrict__ xl, const int* __restrict__ ei,
                         const float* __restrict__ expv, const float* __restrict__ denom,
                         float* __restrict__ gout) {
  int e = blockIdx.x * 8 + (threadIdx.x >> 5);
  if (e >= EN) return;
  int lane = threadIdx.x & 31;
  int src, dst;
  if (e < E_EDGES) { src = ei[e]; dst = ei[E_EDGES + e]; if (src == dst) return; }
  else             { src = dst = e - E_EDGES; }
  int h = lane >> 3;
  float alpha = expv[e * HEADS + h] / denom[dst * HEADS + h];
  const float4* a = (const float4*)(xl + (size_t)src * HF) + lane * 32;
  float* o = gout + (size_t)dst * HF + lane * 128;
#pragma unroll 4
  for (int i = 0; i < 32; ++i) {
    float4 v = a[i];
    atomicAdd(o + i * 4 + 0, alpha * v.x);
    atomicAdd(o + i * 4 + 1, alpha * v.y);
    atomicAdd(o + i * 4 + 2, alpha * v.z);
    atomicAdd(o + i * 4 + 3, alpha * v.w);
  }
}

// ---------------- fc2: 1x1 conv 64->16 ----------------
__global__ void fc2_kernel(const float* __restrict__ g, const float* __restrict__ w2,
                           const float* __restrict__ b2, float* __restrict__ y2) {
  __shared__ float ws[C_CH * 64];
  __shared__ float bs[C_CH];
  int n = blockIdx.x, t = threadIdx.x;
  for (int i = t; i < C_CH * 64; i += 64) ws[i] = w2[i];
  if (t < C_CH) bs[t] = b2[t];
  __syncthreads();
  const float* gp = g + (size_t)n * HF + t;
  float gv[64];
#pragma unroll
  for (int c = 0; c < 64; ++c) gv[c] = gp[c * PX];
  float* yp = y2 + (size_t)n * F_DIM + t;
#pragma unroll
  for (int o = 0; o < C_CH; ++o) {
    float a = bs[o];
#pragma unroll
    for (int c = 0; c < 64; ++c) a = fmaf(ws[o * 64 + c], gv[c], a);
    yp[o * PX] = a;
  }
}

// ---------------- BN2 + residual ----------------
__global__ void final_kernel(const float* __restrict__ y2, const float* __restrict__ scale,
                             const float* __restrict__ shift, const float* __restrict__ x,
                             float* __restrict__ out) {
  int i = blockIdx.x * 256 + threadIdx.x;     // exactly N*F threads
  int c = (i >> 6) & 15;
  out[i] = fmaf(y2[i], scale[c], shift[c]) + x[i];
}

// ---------------- host launcher ----------------
extern "C" void kernel_launch(void* const* d_in, const int* in_sizes, int n_in,
                              void* d_out, int out_size, void* d_ws, size_t ws_size,
                              hipStream_t stream) {
  (void)in_sizes; (void)n_in; (void)out_size; (void)ws_size;
  const float* x    = (const float*)d_in[0];
  const int*   ei   = (const int*)d_in[1];
  const float* w1   = (const float*)d_in[2];
  const float* b1   = (const float*)d_in[3];
  const float* g1   = (const float*)d_in[4];
  const float* be1  = (const float*)d_in[5];
  const float* wl   = (const float*)d_in[6];
  const float* bl   = (const float*)d_in[7];
  const float* wr   = (const float*)d_in[8];
  const float* br   = (const float*)d_in[9];
  const float* att  = (const float*)d_in[10];
  const float* bgat = (const float*)d_in[11];
  const float* w2   = (const float*)d_in[12];
  const float* b2   = (const float*)d_in[13];
  const float* g2   = (const float*)d_in[14];
  const float* be2  = (const float*)d_in[15];
  float* out = (float*)d_out;

  char* wsb = (char*)d_ws;
  size_t off = 0;
  auto alloc = [&](size_t bytes) -> void* {
    void* p = wsb + off;
    off = (off + bytes + 255) & ~(size_t)255;
    return p;
  };
  float*          y1     = (float*)alloc(sizeof(float) * N_NODES * F_DIM);           // 8 MB
  unsigned short* yfb    = (unsigned short*)alloc(2ull * N_NODES * F_DIM);           // 4 MB
  unsigned short* wlb    = (unsigned short*)alloc(2ull * F_DIM * HF);                // 8 MB
  unsigned short* wrb    = (unsigned short*)alloc(2ull * F_DIM * HF);                // 8 MB
  float*          xl     = (float*)alloc(sizeof(float) * N_NODES * HF);              // 32 MB
  float*          xr     = (float*)alloc(sizeof(float) * N_NODES * HF);              // 32 MB
  float*          gout   = (float*)alloc(sizeof(float) * N_NODES * HF);              // 32 MB
  float*          logits = (float*)alloc(sizeof(float) * EN * HEADS);
  float*          expv   = (float*)alloc(sizeof(float) * EN * HEADS);
  float*          mx     = (float*)alloc(sizeof(float) * N_NODES * HEADS);
  float*          denom  = (float*)alloc(sizeof(float) * N_NODES * HEADS);
  float*          y2     = (float*)alloc(sizeof(float) * N_NODES * F_DIM);           // 8 MB
  float*          scale1 = (float*)alloc(sizeof(float) * C_CH);
  float*          shift1 = (float*)alloc(sizeof(float) * C_CH);
  float*          scale2 = (float*)alloc(sizeof(float) * C_CH);
  float*          shift2 = (float*)alloc(sizeof(float) * C_CH);

  // weights -> bf16 (recomputed every call; deterministic)
  cvt_bf16_kernel<<<(F_DIM * HF + 255) / 256, 256, 0, stream>>>(wl, wlb, F_DIM * HF);
  cvt_bf16_kernel<<<(F_DIM * HF + 255) / 256, 256, 0, stream>>>(wr, wrb, F_DIM * HF);

  // fc1 + BN1
  fc1_kernel<<<N_NODES, 64, 0, stream>>>(x, w1, b1, y1);
  bn_stats_kernel<<<C_CH, 256, 0, stream>>>(y1, g1, be1, scale1, shift1);
  bn_apply_cvt_kernel<<<(N_NODES * F_DIM) / 256, 256, 0, stream>>>(y1, scale1, shift1, yfb);

  // GATv2 linear transforms via bf16 WMMA (+ TDM async tile staging)
  dim3 ggrid(HF / 128, N_NODES / 128);
  gemm_bf16_wmma_kernel<<<ggrid, 256, 0, stream>>>(yfb, wlb, bl, xl);
  gemm_bf16_wmma_kernel<<<ggrid, 256, 0, stream>>>(yfb, wrb, br, xr);

  // edge softmax + aggregation
  gat_init_kernel<<<(N_NODES * HF) / 256, 256, 0, stream>>>(gout, bgat);
  softmax_init_kernel<<<(N_NODES * HEADS + 255) / 256, 256, 0, stream>>>(mx, denom);
  edge_logits_kernel<<<(EN + 7) / 8, 256, 0, stream>>>(xl, xr, att, ei, logits, mx);
  edge_expsum_kernel<<<(EN * HEADS + 255) / 256, 256, 0, stream>>>(logits, ei, mx, denom, expv);
  edge_scatter_kernel<<<(EN + 7) / 8, 256, 0, stream>>>(xl, ei, expv, denom, gout);

  // fc2 + BN2 + residual
  fc2_kernel<<<N_NODES, 64, 0, stream>>>(gout, w2, b2, y2);
  bn_stats_kernel<<<C_CH, 256, 0, stream>>>(y2, g2, be2, scale2, shift2);
  final_kernel<<<(N_NODES * F_DIM) / 256, 256, 0, stream>>>(y2, scale2, shift2, x, out);
}
